// MixtureOfExperts_24103356465249
// MI455X (gfx1250) — matrix operand order repaired and checked by reading, not compiled, not run on hardware
//
#include <hip/hip_runtime.h>
#include <hip/hip_bf16.h>
#include <math.h>
#include <stdint.h>

// MoE forward: B=4,S=2048,D=1024,H=4096,E=8,topK=2 on gfx1250 (MI455X).
// Compute-bound (~1.1 TFLOP dense vs ~0.5 GB traffic): bf16 WMMA GEMMs with
// f32 accumulation. Weights are converted f32->bf16 and pre-transposed ONCE
// per expert (memory-bound, trivial), so the GEMM hot loop stages tiles with
// async LDS copies (global_load_async_to_lds_b128, ASYNCcnt) and runs
// 8 v_wmma_f32_16x16x32_bf16 per wave per K-step, double-buffered.

typedef __attribute__((ext_vector_type(16))) __bf16 bf16x16;
typedef __attribute__((ext_vector_type(8)))  __bf16 bf16x8;
typedef __attribute__((ext_vector_type(8)))  float  f32x8;

#define T_TOK 8192
#define DIM   1024
#define HID   4096
#define NEXP  8

#define GBM 128
#define GBN 128
#define GBK 32
#define LSTR 40   // LDS row stride in bf16 elems (32 + 8 pad, keeps 16B align)

__device__ __forceinline__ unsigned short f2bf(float f) {
  __bf16 b = (__bf16)f;
  return __builtin_bit_cast(unsigned short, b);
}

__device__ __forceinline__ f32x8 wmma_bf16(bf16x16 a, bf16x16 b, f32x8 c) {
  // (neg_a, A, neg_b, B, c_mod, C, reuse_a, reuse_b)
  return __builtin_amdgcn_wmma_f32_16x16x32_bf16(false, a, false, b, (short)0, c,
                                                 false, false);
}

// A fragment (16-bit 16x32, ISA layout): lane m=l&15, half=l>>4 holds
// K = half*8 + {0..7} and K = 16 + half*8 + {0..7}.
__device__ __forceinline__ bf16x16 frag_a(const __bf16* lds, int mt, int lane) {
  const int m = lane & 15, h = lane >> 4;
  const __bf16* p = lds + (mt + m) * LSTR + h * 8;
  union { bf16x16 v; bf16x8 q[2]; } u;
  u.q[0] = *(const bf16x8*)(p);
  u.q[1] = *(const bf16x8*)(p + 16);
  return u.v;
}

// B fragment (32x16): lane n=l&15, half=l>>4 holds K = half*8 + {0..7} and K = 16 + half*8 + {0..7}.
// B tile is staged [n][k] in LDS so this is two 16-byte loads per lane.
__device__ __forceinline__ bf16x16 frag_b(const __bf16* lds, int nt, int lane) {
  const int n = lane & 15, kh = lane >> 4;
  const __bf16* p = lds + (nt + n) * LSTR + kh * 8;
  union { bf16x16 v; bf16x8 q[2]; } u;
  u.q[0] = *(const bf16x8*)(p);
  u.q[1] = *(const bf16x8*)(p + 16);
  return u.v;
}

// ---------------- Router: one wave (32 lanes) per token ----------------
__global__ void __launch_bounds__(256) moe_router(
    const float* __restrict__ X, const float* __restrict__ Wr,
    float* __restrict__ gate, float* __restrict__ probs) {
  const int tok  = (int)((blockIdx.x * blockDim.x + threadIdx.x) >> 5);
  const int lane = (int)(threadIdx.x & 31);
  if (tok >= T_TOK) return;
  const float* xr = X + (size_t)tok * DIM;

  float acc[NEXP];
#pragma unroll
  for (int e = 0; e < NEXP; e++) acc[e] = 0.0f;
  for (int d = lane; d < DIM; d += 32) {
    const float xv = xr[d];
    const float4 w0 = *(const float4*)(Wr + (size_t)d * NEXP);
    const float4 w1 = *(const float4*)(Wr + (size_t)d * NEXP + 4);
    acc[0] += xv * w0.x; acc[1] += xv * w0.y; acc[2] += xv * w0.z; acc[3] += xv * w0.w;
    acc[4] += xv * w1.x; acc[5] += xv * w1.y; acc[6] += xv * w1.z; acc[7] += xv * w1.w;
  }
#pragma unroll
  for (int e = 0; e < NEXP; e++) {
#pragma unroll
    for (int off = 16; off > 0; off >>= 1)
      acc[e] += __shfl_xor(acc[e], off, 32);
  }
  if (lane == 0) {
    float mx = acc[0];
#pragma unroll
    for (int e = 1; e < NEXP; e++) mx = fmaxf(mx, acc[e]);
    float p[NEXP]; float s = 0.0f;
#pragma unroll
    for (int e = 0; e < NEXP; e++) { p[e] = expf(acc[e] - mx); s += p[e]; }
    const float inv = 1.0f / s;
#pragma unroll
    for (int e = 0; e < NEXP; e++) p[e] *= inv;
    int i1 = 0;
#pragma unroll
    for (int e = 1; e < NEXP; e++) if (p[e] > p[i1]) i1 = e;
    int i2 = (i1 == 0) ? 1 : 0;
#pragma unroll
    for (int e = 0; e < NEXP; e++) if (e != i1 && p[e] > p[i2]) i2 = e;
    const float wn = 1.0f / (p[i1] + p[i2]);
#pragma unroll
    for (int e = 0; e < NEXP; e++) {
      probs[(size_t)tok * NEXP + e] = p[e];
      gate[(size_t)tok * NEXP + e] =
          (e == i1) ? p[i1] * wn : ((e == i2) ? p[i2] * wn : 0.0f);
    }
  }
}

// ------------- Elementwise f32 -> bf16 (for x), 8 elems/thread -------------
__global__ void __launch_bounds__(256) cvt_f32_to_bf16(
    const float* __restrict__ S, unsigned short* __restrict__ D, int n) {
  const int i = (int)(blockIdx.x * 256 + threadIdx.x) * 8;
  if (i >= n) return;
  const float4 a = *(const float4*)(S + i);
  const float4 b = *(const float4*)(S + i + 4);
  union { unsigned short s[8]; uint4 v; } o;
  o.s[0] = f2bf(a.x); o.s[1] = f2bf(a.y); o.s[2] = f2bf(a.z); o.s[3] = f2bf(a.w);
  o.s[4] = f2bf(b.x); o.s[5] = f2bf(b.y); o.s[6] = f2bf(b.z); o.s[7] = f2bf(b.w);
  *(uint4*)(D + i) = o.v;
}

// ------ Tiled transpose + convert: f32 [R][C] -> bf16 [C][R] ------
__global__ void __launch_bounds__(256) transpose_to_bf16t(
    const float* __restrict__ S, unsigned short* __restrict__ D, int R, int C) {
  __shared__ float tile[32][33];
  const int c0 = blockIdx.x * 32, r0 = blockIdx.y * 32;
  const int tid = (int)threadIdx.x;
  {
    const int r = tid >> 3, c4 = (tid & 7) * 4;
    const float4 v = *(const float4*)(S + (size_t)(r0 + r) * C + c0 + c4);
    tile[r][c4 + 0] = v.x; tile[r][c4 + 1] = v.y;
    tile[r][c4 + 2] = v.z; tile[r][c4 + 3] = v.w;
  }
  __syncthreads();
  {
    const int c = tid >> 3, r4 = (tid & 7) * 4;
    union { unsigned short s[4]; uint2 u; } o;
    o.s[0] = f2bf(tile[r4 + 0][c]); o.s[1] = f2bf(tile[r4 + 1][c]);
    o.s[2] = f2bf(tile[r4 + 2][c]); o.s[3] = f2bf(tile[r4 + 3][c]);
    *(uint2*)(D + (size_t)(c0 + c) * R + r0 + r4) = o.u;
  }
}

// ---------------- Unified bf16 WMMA GEMM ----------------
// C[m,n] = sum_k A[m,k] * Bt[n,k];  A: [M][K] bf16, Bt: [N][K] bf16.
// mode 0: Hout[m,n] = bf16(gelu(C))     (exact erf gelu)
// mode 1: Out[m,n] (+)= gate[m,expert] * C
// Staging uses async LDS copies (ASYNCcnt) with double buffering.
__global__ void __launch_bounds__(256) moe_gemm_bf16(
    const unsigned short* __restrict__ A, const unsigned short* __restrict__ Bt,
    int M, int N, int K, int mode,
    unsigned short* __restrict__ Hout,
    const float* __restrict__ gate, int expert, int accum,
    float* __restrict__ Out) {
  __shared__ alignas(16) __bf16 As[2][GBM * LSTR];
  __shared__ alignas(16) __bf16 Bs[2][GBN * LSTR];
  const int n0 = blockIdx.x * GBN;
  const int m0 = blockIdx.y * GBM;
  const int tid  = (int)threadIdx.x;
  const int lane = tid & 31, wv = tid >> 5;
  const int wm = wv >> 1, wn = wv & 1;  // wave tile origin (wm*32, wn*64)

  f32x8 acc[2][4] = {};

  // staging map: 256 threads cover 128 rows x 32 k, 16 bf16 (32B) per thread
  const int srow = tid >> 1, soff = (tid & 1) * 16;
  const unsigned short* ga0 = A  + (size_t)(m0 + srow) * K + soff;
  const unsigned short* gb0 = Bt + (size_t)(n0 + srow) * K + soff;

#define STAGE(k0_, b_)                                                         \
  do {                                                                         \
    unsigned la_ = (unsigned)(uintptr_t)&As[(b_)][srow * LSTR + soff];         \
    unsigned lb_ = (unsigned)(uintptr_t)&Bs[(b_)][srow * LSTR + soff];         \
    const unsigned short* ga_ = ga0 + (k0_);                                   \
    const unsigned short* gb_ = gb0 + (k0_);                                   \
    asm volatile("global_load_async_to_lds_b128 %0, %1, off"                   \
                 :: "v"(la_), "v"(ga_) : "memory");                            \
    asm volatile("global_load_async_to_lds_b128 %0, %1, off offset:16"         \
                 :: "v"(la_), "v"(ga_) : "memory");                            \
    asm volatile("global_load_async_to_lds_b128 %0, %1, off"                   \
                 :: "v"(lb_), "v"(gb_) : "memory");                            \
    asm volatile("global_load_async_to_lds_b128 %0, %1, off offset:16"         \
                 :: "v"(lb_), "v"(gb_) : "memory");                            \
  } while (0)

  STAGE(0, 0);
  int cur = 0;
  for (int k0 = 0; k0 < K; k0 += GBK) {
    asm volatile("s_wait_asynccnt 0x0" ::: "memory");
    __syncthreads();  // all waves' async copies for buffer `cur` have landed
    const bf16x16 a0f = frag_a(As[cur], wm * 32,      lane);
    const bf16x16 a1f = frag_a(As[cur], wm * 32 + 16, lane);
    bf16x16 bfr[4];
#pragma unroll
    for (int j = 0; j < 4; j++) bfr[j] = frag_b(Bs[cur], wn * 64 + j * 16, lane);
    if (k0 + GBK < K) STAGE(k0 + GBK, cur ^ 1);  // overlap with WMMA below
#pragma unroll
    for (int j = 0; j < 4; j++) {
      acc[0][j] = wmma_bf16(a0f, bfr[j], acc[0][j]);
      acc[1][j] = wmma_bf16(a1f, bfr[j], acc[1][j]);
    }
    cur ^= 1;
  }
#undef STAGE

  // epilogue; C/D layout: lane n=l&15, rows (l>>4)*8 + r
  const int ln = lane & 15, lh = lane >> 4;
#pragma unroll
  for (int i = 0; i < 2; i++) {
    const int tm = m0 + wm * 32 + i * 16 + lh * 8;
    float gv[8];
    if (mode == 1) {
#pragma unroll
      for (int r = 0; r < 8; r++)
        gv[r] = gate[(size_t)(tm + r) * NEXP + expert];
    }
#pragma unroll
    for (int j = 0; j < 4; j++) {
      const int tn = n0 + wn * 64 + j * 16 + ln;
#pragma unroll
      for (int r = 0; r < 8; r++) {
        const float v = acc[i][j][r];
        if (mode == 0) {
          const float g = 0.5f * v * (1.0f + erff(v * 0.70710678118654752f));
          Hout[(size_t)(tm + r) * N + tn] = f2bf(g);
        } else {
          const size_t o = (size_t)(tm + r) * N + tn;
          const float y = gv[r] * v;
          Out[o] = accum ? (Out[o] + y) : y;
        }
      }
    }
  }
}

// --------- Aux loss: deterministic tree reduction over probs ----------
__global__ void __launch_bounds__(256) moe_aux(const float* __restrict__ probs,
                                               float* __restrict__ out_aux) {
  __shared__ float red[256 * NEXP];
  const int tid = (int)threadIdx.x;
  float s[NEXP];
#pragma unroll
  for (int e = 0; e < NEXP; e++) s[e] = 0.0f;
  for (int t = tid; t < T_TOK; t += 256) {
#pragma unroll
    for (int e = 0; e < NEXP; e++) s[e] += probs[(size_t)t * NEXP + e];
  }
#pragma unroll
  for (int e = 0; e < NEXP; e++) red[tid * NEXP + e] = s[e];
  __syncthreads();
  for (int off = 128; off > 0; off >>= 1) {
    if (tid < off) {
#pragma unroll
      for (int e = 0; e < NEXP; e++)
        red[tid * NEXP + e] += red[(tid + off) * NEXP + e];
    }
    __syncthreads();
  }
  if (tid == 0) {
    float loss = 0.0f;
#pragma unroll
    for (int e = 0; e < NEXP; e++) {
      const float m = red[e] / (float)T_TOK - (1.0f / (float)NEXP);
      loss += m * m;
    }
    out_aux[0] = loss;
  }
}

extern "C" void kernel_launch(void* const* d_in, const int* in_sizes, int n_in,
                              void* d_out, int out_size, void* d_ws, size_t ws_size,
                              hipStream_t stream) {
  const float* X  = (const float*)d_in[0];  // [T, D]
  const float* Wr = (const float*)d_in[1];  // [D, E]
  const float* W1 = (const float*)d_in[2];  // [E, D, H]
  const float* W2 = (const float*)d_in[3];  // [E, H, D]
  float* Out = (float*)d_out;               // [T*D] out + [1] aux

  // workspace layout (~96.5 MB)
  char* ws = (char*)d_ws;
  size_t off = 0;
  float* gate  = (float*)(ws + off); off += (size_t)T_TOK * NEXP * 4;
  float* probs = (float*)(ws + off); off += (size_t)T_TOK * NEXP * 4;
  unsigned short* Xb  = (unsigned short*)(ws + off); off += (size_t)T_TOK * DIM * 2;
  unsigned short* Hb  = (unsigned short*)(ws + off); off += (size_t)T_TOK * HID * 2;
  unsigned short* W1t = (unsigned short*)(ws + off); off += (size_t)DIM * HID * 2;
  unsigned short* W2t = (unsigned short*)(ws + off); off += (size_t)HID * DIM * 2;

  moe_router<<<T_TOK / 8, 256, 0, stream>>>(X, Wr, gate, probs);
  cvt_f32_to_bf16<<<(T_TOK * DIM) / (256 * 8), 256, 0, stream>>>(X, Xb,
                                                                 T_TOK * DIM);
  for (int e = 0; e < NEXP; e++) {
    const float* W1e = W1 + (size_t)e * DIM * HID;  // [D][H]
    const float* W2e = W2 + (size_t)e * HID * DIM;  // [H][D]
    // W1t: [H][D]  (B^T for GEMM1)
    transpose_to_bf16t<<<dim3(HID / 32, DIM / 32), 256, 0, stream>>>(
        W1e, W1t, DIM, HID);
    moe_gemm_bf16<<<dim3(HID / GBN, T_TOK / GBM), 256, 0, stream>>>(
        Xb, W1t, T_TOK, HID, DIM, /*mode=*/0, Hb, nullptr, 0, 0, nullptr);
    // W2t: [D][H]  (B^T for GEMM2)
    transpose_to_bf16t<<<dim3(DIM / 32, HID / 32), 256, 0, stream>>>(
        W2e, W2t, HID, DIM);
    moe_gemm_bf16<<<dim3(DIM / GBN, T_TOK / GBM), 256, 0, stream>>>(
        Hb, W2t, T_TOK, DIM, HID, /*mode=*/1, nullptr, gate, e, e > 0, Out);
  }
  moe_aux<<<1, 256, 0, stream>>>(probs, Out + (size_t)T_TOK * DIM);
}
